// LETTER_56384330662372
// MI455X (gfx1250) — compile-verified, matrix operand order checked by reading, not run on hardware
//
#include <hip/hip_runtime.h>
#include <hip/hip_bf16.h>

typedef __attribute__((ext_vector_type(16))) _Float16 v16h;
typedef __attribute__((ext_vector_type(8)))  _Float16 v8h;
typedef __attribute__((ext_vector_type(8)))  float    v8f;
typedef __attribute__((ext_vector_type(4)))  float    v4f;
typedef __attribute__((ext_vector_type(4)))  int      v4i;

#define BDIM 65536
#define DIN 768
#define DCODE 128
#define NLVL 4
#define KCB 1024
#define HDIM 256

#define AS1 __attribute__((address_space(1)))
#define AS3 __attribute__((address_space(3)))

#if __has_builtin(__builtin_amdgcn_global_load_async_to_lds_b128) && \
    __has_builtin(__builtin_amdgcn_s_wait_asynccnt)
#define ASYNC_LDS 1
#else
#define ASYNC_LDS 0
#endif

// stage 16 bytes per lane from global into LDS (ASYNCcnt path when available)
__device__ __forceinline__ void stage16B(const _Float16* g, _Float16* l) {
#if ASYNC_LDS
    __builtin_amdgcn_global_load_async_to_lds_b128((AS1 v4i*)(void*)g,
                                                   (AS3 v4i*)(void*)l, 0, 0);
#else
    v4i t = *(const v4i*)g;
    *(v4i*)l = t;
#endif
}

__device__ __forceinline__ void stage_fence() {
#if ASYNC_LDS
    __builtin_amdgcn_s_wait_asynccnt(0);
#endif
    __syncthreads();
}

// ---------------- tile loaders (ISA 7.12.2 16-bit A/B layout) ----------------
// lane L: row = L&15, K-half g = L>>4.
// v16h elems 0..7  -> K = g*8 + 0..7 ; elems 8..15 -> K = 16 + g*8 + 0..7
__device__ __forceinline__ v16h load_tile(const _Float16* src, int ld, int lane) {
    int r = lane & 15, g = lane >> 4;
    const _Float16* base = src + (size_t)r * ld + g * 8;
    v8h lo = *(const v8h*)(base);
    v8h hi = *(const v8h*)(base + 16);
    v16h out;
#pragma unroll
    for (int i = 0; i < 8; ++i) { out[i] = lo[i]; out[8 + i] = hi[i]; }
    return out;
}

__device__ __forceinline__ v16h load_tile(const float* src, int ld, int lane) {
    int r = lane & 15, g = lane >> 4;
    const float* base = src + (size_t)r * ld + g * 8;
    v4f a0 = *(const v4f*)(base);
    v4f a1 = *(const v4f*)(base + 4);
    v4f b0 = *(const v4f*)(base + 16);
    v4f b1 = *(const v4f*)(base + 20);
    v16h out;
#pragma unroll
    for (int i = 0; i < 4; ++i) {
        out[i]      = (_Float16)a0[i];
        out[4 + i]  = (_Float16)a1[i];
        out[8 + i]  = (_Float16)b0[i];
        out[12 + i] = (_Float16)b1[i];
    }
    return out;
}

#define WMMA(a, b, c) __builtin_amdgcn_wmma_f32_16x16x32_f16(false, a, false, b, (short)0, c, false, false)

// ---------------- generic fused GEMM:  out = act(A @ W^T + bias) ----------------
// A: M x K (f32 or f16), W: N x K (f16). One wave = 16(M) x 64(N) tile.
// All 8 waves of a block share one N-strip (requires M % 128 == 0): the 64x32
// weight slab is staged once per block into LDS (double-buffered, async).
template <typename AT, bool RELU, bool OUT_HALF, bool LOSS>
__global__ __launch_bounds__(256) void gemm_wmma(
    const AT* __restrict__ A, const _Float16* __restrict__ W,
    const float* __restrict__ bias, void* __restrict__ out,
    const float* __restrict__ xref, float* __restrict__ accum,
    int M, int N, int K)
{
    __shared__ _Float16 wtiles[2][4 * 16 * 40];   // [buf][subtile][row][k(+pad)]
    int tid = threadIdx.x;
    int lane = tid & 31;
    int unit = blockIdx.x * 8 + (tid >> 5);
    int mdiv = M >> 4;
    int m16   = unit % mdiv;
    int strip = unit / mdiv;              // block-uniform (mdiv % 8 == 0)
    int mrow = m16 << 4;
    int colbase = strip << 6;

    // staging map: this thread moves 8 halves (16B) of the 64x32 slab
    int t_t = tid >> 6;                   // subtile 0..3 (16 N-rows each)
    int t_r = (tid & 63) >> 2;            // row within subtile
    int t_k = (tid & 3) * 8;              // k offset
    const _Float16* gsrc = W + (size_t)(colbase + t_t * 16 + t_r) * K + t_k;
    int ldst = t_t * 640 + t_r * 40 + t_k;

    stage16B(gsrc, &wtiles[0][ldst]);
    stage_fence();

    v8f acc0 = 0.f, acc1 = 0.f, acc2 = 0.f, acc3 = 0.f;
    const int nk = K >> 5;
    for (int i = 0; i < nk; ++i) {
        int k0 = i << 5;
        if (i + 1 < nk) {
            stage16B(gsrc + k0 + 32, &wtiles[(i + 1) & 1][ldst]);
            __builtin_prefetch(A + (size_t)mrow * K + k0 + 32 + (lane & 15) * K, 0, 3);
        }
        v16h a  = load_tile(A + (size_t)mrow * K + k0, K, lane);
        const _Float16* bp = wtiles[i & 1];
        v16h b0 = load_tile(bp + 0 * 640, 40, lane);
        v16h b1 = load_tile(bp + 1 * 640, 40, lane);
        v16h b2 = load_tile(bp + 2 * 640, 40, lane);
        v16h b3 = load_tile(bp + 3 * 640, 40, lane);
        acc0 = WMMA(a, b0, acc0);
        acc1 = WMMA(a, b1, acc1);
        acc2 = WMMA(a, b2, acc2);
        acc3 = WMMA(a, b3, acc3);
        stage_fence();                    // WMMAs consumed b* -> dscnt already waited
    }

    int r = lane & 15, g = lane >> 4;
    float lsum = 0.f;
    v8f accs[4] = {acc0, acc1, acc2, acc3};
#pragma unroll
    for (int t = 0; t < 4; ++t) {
        int ncol = colbase + t * 16 + r;
        float bv = bias[ncol];
#pragma unroll
        for (int i = 0; i < 8; ++i) {
            int m = mrow + i + 8 * g;     // C/D layout: VGPR i, half g
            float v = accs[t][i] + bv;
            if (RELU) v = fmaxf(v, 0.f);
            size_t off = (size_t)m * N + ncol;
            if (OUT_HALF) {
                ((_Float16*)out)[off] = (_Float16)v;
            } else {
                ((float*)out)[off] = v;
                if (LOSS) { float d = v - xref[off]; lsum += d * d; }
            }
        }
    }
    if (LOSS) {
#pragma unroll
        for (int msk = 16; msk >= 1; msk >>= 1) lsum += __shfl_xor(lsum, msk, 32);
        if (lane == 0) atomicAdd(accum, lsum);
    }
}

// ---------------- per-level RQ: distances + argmin + gather + residual update ----
// Block = 8 waves x 16 rows. All waves sweep the same 1024-code codebook:
// 16x128 codebook slabs staged through LDS (double-buffered, async).
__global__ __launch_bounds__(256) void rq_level(
    float* __restrict__ residual, const _Float16* __restrict__ cbh,
    const float* __restrict__ cbf, const float* __restrict__ enorm,
    float* __restrict__ codes, float* __restrict__ accum, int level)
{
    __shared__ _Float16 tiles[2][16 * 136];       // padded ld=136
    __shared__ int s_idx[8][16];
    int tid = threadIdx.x;
    int lane = tid & 31, w = tid >> 5;
    int mrow = (blockIdx.x * 8 + w) << 4;
    int r = lane & 15, g = lane >> 4;

    // staging map: thread moves 8 halves of the 16x128 slab
    int s_row = tid >> 4;
    int s_k   = (tid & 15) * 8;
    int ldst = s_row * 136 + s_k;
    const _Float16* gsrc = cbh + (size_t)s_row * DCODE + s_k;

    // residual A-tiles for K=128 held in registers across the whole sweep
    const float* ra = residual + (size_t)mrow * DCODE;
    v16h a0 = load_tile(ra +  0, DCODE, lane);
    v16h a1 = load_tile(ra + 32, DCODE, lane);
    v16h a2 = load_tile(ra + 64, DCODE, lane);
    v16h a3 = load_tile(ra + 96, DCODE, lane);

    stage16B(gsrc, &tiles[0][ldst]);
    stage_fence();

    float best[8];
    int bidx[8];
#pragma unroll
    for (int i = 0; i < 8; ++i) { best[i] = 3.0e38f; bidx[i] = 0; }

    for (int nt = 0; nt < KCB / 16; ++nt) {
        if (nt + 1 < KCB / 16)
            stage16B(gsrc + (size_t)(nt + 1) * 16 * DCODE, &tiles[(nt + 1) & 1][ldst]);
        const _Float16* bp = tiles[nt & 1];
        v16h b0 = load_tile(bp +  0, 136, lane);
        v16h b1 = load_tile(bp + 32, 136, lane);
        v16h b2 = load_tile(bp + 64, 136, lane);
        v16h b3 = load_tile(bp + 96, 136, lane);
        v8f s = 0.f;
        s = WMMA(a0, b0, s);
        s = WMMA(a1, b1, s);
        s = WMMA(a2, b2, s);
        s = WMMA(a3, b3, s);
        int n = nt * 16 + r;
        float en = enorm[n];              // ||r||^2 is row-constant: argmin-invariant
#pragma unroll
        for (int i = 0; i < 8; ++i) {
            float d = en - 2.0f * s[i];
            if (d < best[i]) { best[i] = d; bidx[i] = n; }
        }
        stage_fence();
    }

    // reduce across the 16 lanes of each half (xor masks stay within halves)
#pragma unroll
    for (int msk = 1; msk <= 8; msk <<= 1) {
#pragma unroll
        for (int i = 0; i < 8; ++i) {
            float ov = __shfl_xor(best[i], msk, 32);
            int   oi = __shfl_xor(bidx[i], msk, 32);
            if (ov < best[i] || (ov == best[i] && oi < bidx[i])) { best[i] = ov; bidx[i] = oi; }
        }
    }
    if (r == 0) {
#pragma unroll
        for (int i = 0; i < 8; ++i) {
            int m = i + 8 * g;
            s_idx[w][m] = bidx[i];
            codes[(size_t)(mrow + m) * NLVL + level] = (float)bidx[i];
        }
    }
    __syncthreads();

    // gather winning code rows (f32), accumulate loss, update residual in place
    float lsum = 0.f;
    for (int m = 0; m < 16; ++m) {
        int idx = s_idx[w][m];
        const float* q = cbf + (size_t)idx * DCODE;
        float* rr = residual + (size_t)(mrow + m) * DCODE;
#pragma unroll
        for (int j = 0; j < 4; ++j) {
            int d = lane * 4 + j;
            float qv = q[d], rv = rr[d];
            float df = qv - rv;
            lsum += df * df;
            rr[d] = rv - qv;
        }
    }
#pragma unroll
    for (int msk = 16; msk >= 1; msk >>= 1) lsum += __shfl_xor(lsum, msk, 32);
    if (lane == 0) atomicAdd(accum, lsum);
}

// ---------------- small helper kernels ----------------
__global__ void zero_accum(float* a) { a[0] = 0.f; a[1] = 0.f; }

__global__ void cvt_f16(const float* __restrict__ src, _Float16* __restrict__ dst, int n) {
    int i = blockIdx.x * blockDim.x + threadIdx.x;
    if (i < n) dst[i] = (_Float16)src[i];
}

__global__ void enorm_k(const float* __restrict__ cb, float* __restrict__ en, int n) {
    int i = blockIdx.x * blockDim.x + threadIdx.x;
    if (i < n) {
        const float* p = cb + (size_t)i * DCODE;
        float s = 0.f;
#pragma unroll 8
        for (int j = 0; j < DCODE; ++j) s += p[j] * p[j];
        en[i] = s;
    }
}

__global__ void copy_f32(float* __restrict__ dst, const float* __restrict__ src, int n) {
    int i = blockIdx.x * blockDim.x + threadIdx.x;
    if (i < n) dst[i] = src[i];
}

__global__ void sub_f32(float* __restrict__ dst, const float* __restrict__ src, int n) {
    int i = blockIdx.x * blockDim.x + threadIdx.x;
    if (i < n) dst[i] -= src[i];
}

__global__ void finalize(const float* __restrict__ accum, float* __restrict__ outs) {
    float rq  = accum[0] * 1.25f / ((float)BDIM * (float)DCODE);  // cb + 0.25*commit
    float rec = accum[1] / ((float)BDIM * (float)DIN);
    outs[0] = rq + rec;
    outs[1] = rq;
    outs[2] = rec;
}

extern "C" void kernel_launch(void* const* d_in, const int* in_sizes, int n_in,
                              void* d_out, int out_size, void* d_ws, size_t ws_size,
                              hipStream_t stream) {
    const float* x      = (const float*)d_in[0];
    const float* enc_w1 = (const float*)d_in[1];
    const float* enc_b1 = (const float*)d_in[2];
    const float* enc_w2 = (const float*)d_in[3];
    const float* enc_b2 = (const float*)d_in[4];
    const float* cbf    = (const float*)d_in[5];
    const float* dec_w1 = (const float*)d_in[6];
    const float* dec_b1 = (const float*)d_in[7];
    const float* dec_w2 = (const float*)d_in[8];
    const float* dec_b2 = (const float*)d_in[9];

    // ---- workspace layout (bytes) ----
    char* ws = (char*)d_ws;
    _Float16* w1h  = (_Float16*)(ws);                       // 256*768
    _Float16* w2h  = (_Float16*)(ws + 393216);              // 128*256
    _Float16* cbh  = (_Float16*)(ws + 458752);              // 4*1024*128
    _Float16* dw1h = (_Float16*)(ws + 1507328);             // 256*128
    _Float16* dw2h = (_Float16*)(ws + 1572864);             // 768*256
    float*    en   = (float*)(ws + 1966080);                // 4*1024
    float*    acc  = (float*)(ws + 1982464);                // 2 floats
    _Float16* hbuf = (_Float16*)(ws + 1982720);             // 65536*256 f16
    float*    res  = (float*)(ws + 35537152);               // 65536*128 f32
    _Float16* hd   = (_Float16*)(ws + 69091584);            // 65536*256 f16

    // ---- d_out layout ----
    float* recon = (float*)d_out;                           // B*768
    float* zq    = recon + (size_t)BDIM * DIN;              // B*128
    float* codes = zq + (size_t)BDIM * DCODE;               // B*4 (indices as float)
    float* outs  = codes + (size_t)BDIM * NLVL;             // total, L_RQVAE, L_Recon

    zero_accum<<<1, 1, 0, stream>>>(acc);

    cvt_f16<<<(HDIM * DIN + 255) / 256, 256, 0, stream>>>(enc_w1, w1h, HDIM * DIN);
    cvt_f16<<<(DCODE * HDIM + 255) / 256, 256, 0, stream>>>(enc_w2, w2h, DCODE * HDIM);
    cvt_f16<<<(NLVL * KCB * DCODE + 255) / 256, 256, 0, stream>>>(cbf, cbh, NLVL * KCB * DCODE);
    cvt_f16<<<(HDIM * DCODE + 255) / 256, 256, 0, stream>>>(dec_w1, dw1h, HDIM * DCODE);
    cvt_f16<<<(DIN * HDIM + 255) / 256, 256, 0, stream>>>(dec_w2, dw2h, DIN * HDIM);
    enorm_k<<<(NLVL * KCB + 255) / 256, 256, 0, stream>>>(cbf, en, NLVL * KCB);

    // encoder: h = relu(x @ W1^T + b1)   [M=65536, N=256, K=768]
    gemm_wmma<float, true, true, false><<<(BDIM / 16) * (HDIM / 64) / 8, 256, 0, stream>>>(
        x, w1h, enc_b1, hbuf, nullptr, nullptr, BDIM, HDIM, DIN);
    // z_e = h @ W2^T + b2  -> residual   [N=128, K=256]
    gemm_wmma<_Float16, false, false, false><<<(BDIM / 16) * (DCODE / 64) / 8, 256, 0, stream>>>(
        hbuf, w2h, enc_b2, res, nullptr, nullptr, BDIM, DCODE, HDIM);

    // save z_e into z_q slot (z_q = z_e - residual_final at the end)
    copy_f32<<<(BDIM * DCODE + 255) / 256, 256, 0, stream>>>(zq, res, BDIM * DCODE);

    // 4 residual-quantization levels (distance WMMA + argmin + gather fused)
    for (int l = 0; l < NLVL; ++l) {
        rq_level<<<BDIM / 16 / 8, 256, 0, stream>>>(
            res, cbh + (size_t)l * KCB * DCODE, cbf + (size_t)l * KCB * DCODE,
            en + (size_t)l * KCB, codes, acc, l);
    }

    // z_q = z_e - residual_final
    sub_f32<<<(BDIM * DCODE + 255) / 256, 256, 0, stream>>>(zq, res, BDIM * DCODE);

    // decoder: hd = relu(z_q @ Wd1^T + bd1)  [N=256, K=128]
    gemm_wmma<float, true, true, false><<<(BDIM / 16) * (HDIM / 64) / 8, 256, 0, stream>>>(
        zq, dw1h, dec_b1, hd, nullptr, nullptr, BDIM, HDIM, DCODE);
    // recon = hd @ Wd2^T + bd2, fused Σ(recon-x)^2   [N=768, K=256]
    gemm_wmma<_Float16, false, false, true><<<(BDIM / 16) * (DIN / 64) / 8, 256, 0, stream>>>(
        hd, dw2h, dec_b2, recon, x, acc + 1, BDIM, DIN, HDIM);

    finalize<<<1, 1, 0, stream>>>(acc, outs);
}
